// DispatchGraphEncoder_39874476376560
// MI455X (gfx1250) — matrix operand order repaired and checked by reading, not compiled, not run on hardware
//
#include <hip/hip_runtime.h>

// ---------------------------------------------------------------------------
// GIN graph encoder for MI455X (gfx1250, wave32, WMMA, TDM).
// Activations kept in bf16 (halves L2/HBM gather traffic), aggregation in
// fp32 hardware atomics. GEMMs via v_wmma_f32_16x16x32_bf16. GIN input tiles
// staged to LDS via the Tensor Data Mover. Activation buffers padded to a
// multiple of 128 rows so all WMMA epilogues are branch-free.
// ---------------------------------------------------------------------------

#define N_NODES  100000
#define M_PAD    100096   // N_NODES rounded up to multiple of 128
#define N_EDGES  800000
#define NODE_DIM 128
#define HIDDEN   256
#define OUT_DIM  512
#define N_LAYERS 4
#define N_GRAPHS 64

typedef __bf16 bf16;
typedef __attribute__((ext_vector_type(16))) __bf16 v16bf;
typedef __attribute__((ext_vector_type(8)))  __bf16 v8bf;
typedef __attribute__((ext_vector_type(8)))  float  v8f;
typedef __attribute__((ext_vector_type(4)))  unsigned int v4u;
typedef __attribute__((ext_vector_type(8)))  int v8i;
typedef __attribute__((ext_vector_type(4)))  int v4i;

// ---------------------------------------------------------------------------
// WMMA helpers
// ---------------------------------------------------------------------------

__device__ __forceinline__ v8f wmma_bf16(v16bf a, v16bf b, v8f c) {
  return __builtin_amdgcn_wmma_f32_16x16x32_bf16(
      /*neg_a=*/false, a, /*neg_b=*/false, b,
      /*c_mod=*/(short)0, c, /*reuse_a=*/false, /*reuse_b=*/false);
}

// A fragment (16x32 bf16, row-major source). ISA 7.12.2 layout:
// lane<16 -> row M=lane, K {0..7} in v0..3, {16..23} in v4..7;
// lane>=16 -> same row, K {8..15},{24..31}.
__device__ __forceinline__ v16bf load_fragA_bf16(const bf16* __restrict__ base,
                                                 int ld, int lane) {
  const int r  = lane & 15;
  const int ko = (lane >> 4) << 3;           // 0 or 8
  const bf16* p = base + (size_t)r * ld + ko;
  v8bf lo = *(const v8bf*)(p);
  v8bf hi = *(const v8bf*)(p + 16);
  v16bf f;
#pragma unroll
  for (int i = 0; i < 8; ++i) { f[i] = lo[i]; f[i + 8] = hi[i]; }
  return f;
}

// A fragment from fp32 source with row clamp (x is exactly N_NODES rows).
__device__ __forceinline__ v16bf load_fragA_f32(const float* __restrict__ base,
                                                int ld, int row0, int rowmax,
                                                int k0, int lane) {
  int r = row0 + (lane & 15);
  r = (r < rowmax) ? r : rowmax;
  const int ko = k0 + ((lane >> 4) << 3);
  const float* p = base + (size_t)r * ld + ko;
  v16bf f;
#pragma unroll
  for (int i = 0; i < 8; ++i) {
    f[i]     = (bf16)p[i];
    f[i + 8] = (bf16)p[i + 16];
  }
  return f;
}

// B fragment (32x16). Source W[N][K] row-major; B column n is W row n.
// lane<16 -> column N=lane, K {0..7},{16..23}; lane>=16 -> K {8..15},{24..31}.
__device__ __forceinline__ v16bf load_fragB_bf16(const bf16* __restrict__ base,
                                                 int ld, int lane) {
  const int n  = lane & 15;
  const int kh = (lane >> 4) << 3;           // 0 or 8
  const bf16* p = base + (size_t)n * ld + kh;
  v8bf lo = *(const v8bf*)(p);
  v8bf hi = *(const v8bf*)(p + 16);
  v16bf f;
#pragma unroll
  for (int i = 0; i < 8; ++i) { f[i] = lo[i]; f[i + 8] = hi[i]; }
  return f;
}

// ---------------------------------------------------------------------------
// Tensor Data Mover: load a [rows=64, cols=256] bf16 tile (row-major,
// stride 256) from global into LDS. D# per CDNA5 ISA 8.3/8.4; 2-D tensor so
// groups 2/3 are zero. Issued by one wave; completion via TENSORcnt.
// This toolchain's builtin takes 6 args (extra zero int32x8 before cpol).
// ---------------------------------------------------------------------------
__device__ __forceinline__ void tdm_load_tile_bf16(const bf16* gsrc,
                                                   unsigned lds_off,
                                                   unsigned rows_total) {
  const unsigned long long ga = (unsigned long long)(size_t)gsrc;
  v4u g0;
  g0[0] = 1u;                                           // count=1 (valid D#)
  g0[1] = lds_off;                                      // lds_addr (bytes)
  g0[2] = (unsigned)(ga & 0xFFFFFFFFu);                 // global_addr[31:0]
  g0[3] = (unsigned)((ga >> 32) & 0x01FFFFFFu)          // global_addr[56:32]
        | (2u << 30);                                   // type = 2 ("image")
  v8i g1;
  g1[0] = 0x10000;                                      // data_size = 2 bytes
  g1[1] = (int)(256u << 16);                            // tensor_dim0 = 256
  g1[2] = (int)((rows_total & 0xFFFFu) << 16);          // tensor_dim1 lo
  g1[3] = (int)((rows_total >> 16) | (256u << 16));     // dim1 hi, tile_dim0=256
  g1[4] = 64;                                           // tile_dim1=64, tile_dim2=0
  g1[5] = 256;                                          // tensor_dim0_stride = 256
  g1[6] = 0;
  g1[7] = 0;
  const v4i z4 = {0, 0, 0, 0};
  const v8i z8 = {0, 0, 0, 0, 0, 0, 0, 0};
  __builtin_amdgcn_tensor_load_to_lds(g0, g1, z4, z4, z8, 0);
}

// ---------------------------------------------------------------------------
// Small utility kernels
// ---------------------------------------------------------------------------

__global__ void cvt_f32_bf16_kernel(const float* __restrict__ src,
                                    bf16* __restrict__ dst, int n) {
  int i = blockIdx.x * blockDim.x + threadIdx.x;
  if (i < n) dst[i] = (bf16)src[i];
}

__global__ void fill_f32_kernel(float* __restrict__ p, float v, int n) {
  int i = blockIdx.x * blockDim.x + threadIdx.x;
  if (i < n) p[i] = v;
}

// ---------------------------------------------------------------------------
// Node projection: h = relu(x @ node_w^T + node_b)  [N,128] x [256,128]^T
// h is padded to M_PAD rows -> store epilogue is branch-free.
// ---------------------------------------------------------------------------
__global__ __launch_bounds__(256)
void node_proj_kernel(const float* __restrict__ x, const bf16* __restrict__ w,
                      const float* __restrict__ b, bf16* __restrict__ h) {
  const int lane = threadIdx.x & 31;
  const int wv   = threadIdx.x >> 5;
  const int m0   = blockIdx.x * 128 + wv * 16;   // grid is exact for M_PAD

  v16bf afr[4];
#pragma unroll
  for (int kk = 0; kk < 4; ++kk)
    afr[kk] = load_fragA_f32(x, NODE_DIM, m0, N_NODES - 1, kk * 32, lane);

  for (int nt = 0; nt < 16; ++nt) {
    const int n0 = nt * 16;
    const float bias = b[n0 + (lane & 15)];
    v8f acc = {bias, bias, bias, bias, bias, bias, bias, bias};
#pragma unroll
    for (int kk = 0; kk < 4; ++kk) {
      v16bf bb = load_fragB_bf16(w + (size_t)n0 * NODE_DIM + kk * 32,
                                 NODE_DIM, lane);
      acc = wmma_bf16(afr[kk], bb, acc);
    }
#pragma unroll
    for (int v = 0; v < 8; ++v) {
      const int row = m0 + v + ((lane >> 4) << 3);
      const int col = n0 + (lane & 15);
      h[(size_t)row * HIDDEN + col] = (bf16)fmaxf(acc[v], 0.0f);
    }
  }
}

// ---------------------------------------------------------------------------
// Edge aggregation: agg[dst] += f32(h[src]); thread = (edge, 8 features).
// ---------------------------------------------------------------------------
__global__ __launch_bounds__(256)
void agg_kernel(const int* __restrict__ src, const int* __restrict__ dst,
                const bf16* __restrict__ h, float* __restrict__ agg) {
  const int idx = blockIdx.x * blockDim.x + threadIdx.x;
  if (idx >= N_EDGES * 32) return;
  const int e = idx >> 5;
  const int c = (idx & 31) << 3;
  const int s = src[e];
  const int d = dst[e];
  v8bf v = *(const v8bf*)(h + (size_t)s * HIDDEN + c);
  float* ap = agg + (size_t)d * HIDDEN + c;
#pragma unroll
  for (int i = 0; i < 8; ++i) unsafeAtomicAdd(ap + i, (float)v[i]);
}

// ---------------------------------------------------------------------------
// Fused GIN MLP layer: z = (1+eps)*h + agg; h' = relu(relu(z@W1^T+b1)@W2^T+b2)
// BM=64 rows per block. h tile DMA'd into LDS by the Tensor Data Mover; z and
// the GEMM1 output live in LDS (bf16); the MLP intermediate never hits HBM.
// LDS = 64KB: zA [0,32K), zB [32K,64K); the TDM staging tile aliases zB
// (only live during phase 0).
// ---------------------------------------------------------------------------
#define BM 64

__global__ __launch_bounds__(256)
void gin_layer_kernel(const bf16* __restrict__ h_in, const float* __restrict__ agg,
                      const float* __restrict__ eps_arr, int layer,
                      const bf16* __restrict__ w1, const float* __restrict__ b1,
                      const bf16* __restrict__ w2, const float* __restrict__ b2,
                      bf16* __restrict__ h_out) {
  __shared__ __align__(16) unsigned char smem[64 * 1024];
  bf16* zA = (bf16*)smem;                  // z tile           [64][256]
  bf16* zB = (bf16*)(smem + 32 * 1024);    // GEMM1 output     [64][256]
  bf16* hT = zB;                           // TDM staging tile (phase 0 only)

  const int tid  = threadIdx.x;
  const int lane = tid & 31;
  const int wv   = tid >> 5;
  const int m0   = blockIdx.x * BM;        // grid is exact for M_PAD
  const float s  = 1.0f + eps_arr[layer];

  // Warm the weight matrices toward the caches (global_prefetch_b8).
  if (tid < 64) {
    __builtin_prefetch(w1 + (size_t)tid * 1024, 0, 0);
    __builtin_prefetch(w2 + (size_t)tid * 1024, 0, 0);
  }

  // TDM: DMA h tile [64][256] bf16 into LDS. One wave issues (TDM ignores
  // EXEC); completion tracked with TENSORcnt, published via the barrier.
  if (wv == 0) {
    tdm_load_tile_bf16(h_in + (size_t)m0 * HIDDEN,
                       (unsigned)(size_t)hT, M_PAD);
    __builtin_amdgcn_s_wait_tensorcnt(0);
  }
  __syncthreads();

  // Phase 0: z = (1+eps)*h + agg  ->  zA (bf16)
  for (int idx = tid; idx < BM * HIDDEN; idx += 256) {
    const size_t off = (size_t)m0 * HIDDEN + idx;
    zA[idx] = (bf16)(s * (float)hT[idx] + agg[off]);
  }
  __syncthreads();

  // Phase 1: zB = relu(zA @ W1^T + b1)   (4x16 tiles, 8 per wave)
  for (int t = wv; t < 64; t += 8) {
    const int mt = t >> 4;
    const int n0 = (t & 15) * 16;
    const float bias = b1[n0 + (lane & 15)];
    v8f acc = {bias, bias, bias, bias, bias, bias, bias, bias};
#pragma unroll
    for (int k = 0; k < HIDDEN; k += 32) {
      v16bf a  = load_fragA_bf16(zA + mt * 16 * HIDDEN + k, HIDDEN, lane);
      v16bf bb = load_fragB_bf16(w1 + (size_t)n0 * HIDDEN + k, HIDDEN, lane);
      acc = wmma_bf16(a, bb, acc);
    }
#pragma unroll
    for (int v = 0; v < 8; ++v) {
      const int rl  = mt * 16 + v + ((lane >> 4) << 3);
      const int col = n0 + (lane & 15);
      zB[rl * HIDDEN + col] = (bf16)fmaxf(acc[v], 0.0f);
    }
  }
  __syncthreads();

  // Phase 2: h_out = relu(zB @ W2^T + b2)  (branch-free store, padded rows)
  for (int t = wv; t < 64; t += 8) {
    const int mt = t >> 4;
    const int n0 = (t & 15) * 16;
    const float bias = b2[n0 + (lane & 15)];
    v8f acc = {bias, bias, bias, bias, bias, bias, bias, bias};
#pragma unroll
    for (int k = 0; k < HIDDEN; k += 32) {
      v16bf a  = load_fragA_bf16(zB + mt * 16 * HIDDEN + k, HIDDEN, lane);
      v16bf bb = load_fragB_bf16(w2 + (size_t)n0 * HIDDEN + k, HIDDEN, lane);
      acc = wmma_bf16(a, bb, acc);
    }
#pragma unroll
    for (int v = 0; v < 8; ++v) {
      const int row = m0 + mt * 16 + v + ((lane >> 4) << 3);
      const int col = n0 + (lane & 15);
      h_out[(size_t)row * HIDDEN + col] = (bf16)fmaxf(acc[v], 0.0f);
    }
  }
}

// ---------------------------------------------------------------------------
// Mean pool: counts + sums via f32 atomics, then mean -> bf16.
// ---------------------------------------------------------------------------
__global__ void count_kernel(const int* __restrict__ batch,
                             float* __restrict__ counts) {
  int n = blockIdx.x * blockDim.x + threadIdx.x;
  if (n < N_NODES) unsafeAtomicAdd(&counts[batch[n]], 1.0f);
}

__global__ __launch_bounds__(256)
void pool_sum_kernel(const bf16* __restrict__ h, const int* __restrict__ batch,
                     float* __restrict__ pooled) {
  const int idx = blockIdx.x * blockDim.x + threadIdx.x;
  if (idx >= N_NODES * 32) return;
  const int n = idx >> 5;
  const int c = (idx & 31) << 3;
  const int g = batch[n];
  v8bf v = *(const v8bf*)(h + (size_t)n * HIDDEN + c);
  float* p = pooled + (size_t)g * HIDDEN + c;
#pragma unroll
  for (int i = 0; i < 8; ++i) unsafeAtomicAdd(p + i, (float)v[i]);
}

__global__ void mean_kernel(const float* __restrict__ pooled,
                            const float* __restrict__ counts,
                            bf16* __restrict__ hg) {
  int i = blockIdx.x * blockDim.x + threadIdx.x;
  if (i < N_GRAPHS * HIDDEN)
    hg[i] = (bf16)(pooled[i] / fmaxf(counts[i >> 8], 1.0f));
}

// ---------------------------------------------------------------------------
// Head: out = relu(hg @ out_w1^T + b1) @ out_w2^T + b2.  Single block.
// ---------------------------------------------------------------------------
__global__ __launch_bounds__(256)
void head_kernel(const bf16* __restrict__ hg, const bf16* __restrict__ ow1,
                 const float* __restrict__ ob1, const bf16* __restrict__ ow2,
                 const float* __restrict__ ob2, float* __restrict__ out) {
  __shared__ bf16 z1[N_GRAPHS * HIDDEN];  // 32 KB
  const int lane = threadIdx.x & 31;
  const int wv   = threadIdx.x >> 5;

  // GEMM1: 64x256 -> 4x16 tiles
  for (int t = wv; t < 64; t += 8) {
    const int mt = t >> 4;
    const int n0 = (t & 15) * 16;
    const float bias = ob1[n0 + (lane & 15)];
    v8f acc = {bias, bias, bias, bias, bias, bias, bias, bias};
#pragma unroll
    for (int k = 0; k < HIDDEN; k += 32) {
      v16bf a  = load_fragA_bf16(hg + (size_t)mt * 16 * HIDDEN + k, HIDDEN, lane);
      v16bf bb = load_fragB_bf16(ow1 + (size_t)n0 * HIDDEN + k, HIDDEN, lane);
      acc = wmma_bf16(a, bb, acc);
    }
#pragma unroll
    for (int v = 0; v < 8; ++v) {
      const int rl  = mt * 16 + v + ((lane >> 4) << 3);
      const int col = n0 + (lane & 15);
      z1[rl * HIDDEN + col] = (bf16)fmaxf(acc[v], 0.0f);
    }
  }
  __syncthreads();

  // GEMM2: 64x512 -> 4x32 tiles, f32 output
  for (int t = wv; t < 128; t += 8) {
    const int mt = t >> 5;
    const int n0 = (t & 31) * 16;
    const float bias = ob2[n0 + (lane & 15)];
    v8f acc = {bias, bias, bias, bias, bias, bias, bias, bias};
#pragma unroll
    for (int k = 0; k < HIDDEN; k += 32) {
      v16bf a  = load_fragA_bf16(z1 + (size_t)mt * 16 * HIDDEN + k, HIDDEN, lane);
      v16bf bb = load_fragB_bf16(ow2 + (size_t)n0 * HIDDEN + k, HIDDEN, lane);
      acc = wmma_bf16(a, bb, acc);
    }
#pragma unroll
    for (int v = 0; v < 8; ++v) {
      const int row = mt * 16 + v + ((lane >> 4) << 3);
      const int col = n0 + (lane & 15);
      out[(size_t)row * OUT_DIM + col] = acc[v];
    }
  }
}

// ---------------------------------------------------------------------------
// Host-side orchestration
// ---------------------------------------------------------------------------
extern "C" void kernel_launch(void* const* d_in, const int* in_sizes, int n_in,
                              void* d_out, int out_size, void* d_ws, size_t ws_size,
                              hipStream_t stream) {
  const float* x      = (const float*)d_in[0];
  const int*   eidx   = (const int*)d_in[1];   // [2, N_EDGES]
  const int*   batch  = (const int*)d_in[2];
  const float* node_w = (const float*)d_in[3];
  const float* node_b = (const float*)d_in[4];
  const float* gin_w1 = (const float*)d_in[5];
  const float* gin_b1 = (const float*)d_in[6];
  const float* gin_w2 = (const float*)d_in[7];
  const float* gin_b2 = (const float*)d_in[8];
  const float* eps    = (const float*)d_in[9];
  const float* out_w1 = (const float*)d_in[10];
  const float* out_b1 = (const float*)d_in[11];
  const float* out_w2 = (const float*)d_in[12];
  const float* out_b2 = (const float*)d_in[13];
  float* out = (float*)d_out;

  // --- workspace carve-up (padded activations) ---
  char* ws = (char*)d_ws;
  size_t off = 0;
  auto take = [&](size_t bytes) -> char* {
    char* p = ws + off;
    off = (off + bytes + 255) & ~(size_t)255;
    return p;
  };
  const size_t NHP = (size_t)M_PAD * HIDDEN;
  bf16*  hA        = (bf16*)take(NHP * sizeof(bf16));
  bf16*  hB        = (bf16*)take(NHP * sizeof(bf16));
  float* agg       = (float*)take(NHP * sizeof(float));
  float* pooled    = (float*)take((size_t)N_GRAPHS * HIDDEN * sizeof(float));
  float* counts    = (float*)take((size_t)N_GRAPHS * sizeof(float));
  bf16*  pooled_bf = (bf16*)take((size_t)N_GRAPHS * HIDDEN * sizeof(bf16));
  bf16*  wnode     = (bf16*)take((size_t)HIDDEN * NODE_DIM * sizeof(bf16));
  bf16*  w1b       = (bf16*)take((size_t)N_LAYERS * HIDDEN * HIDDEN * sizeof(bf16));
  bf16*  w2b       = (bf16*)take((size_t)N_LAYERS * HIDDEN * HIDDEN * sizeof(bf16));
  bf16*  ow1b      = (bf16*)take((size_t)HIDDEN * HIDDEN * sizeof(bf16));
  bf16*  ow2b      = (bf16*)take((size_t)OUT_DIM * HIDDEN * sizeof(bf16));

  // --- weight conversion (f32 -> bf16) ---
  auto cvt = [&](const float* s, bf16* d, int n) {
    cvt_f32_bf16_kernel<<<(n + 255) / 256, 256, 0, stream>>>(s, d, n);
  };
  cvt(node_w, wnode, HIDDEN * NODE_DIM);
  cvt(gin_w1, w1b, N_LAYERS * HIDDEN * HIDDEN);
  cvt(gin_w2, w2b, N_LAYERS * HIDDEN * HIDDEN);
  cvt(out_w1, ow1b, HIDDEN * HIDDEN);
  cvt(out_w2, ow2b, OUT_DIM * HIDDEN);

  // --- node projection (exact grid over padded M) ---
  node_proj_kernel<<<M_PAD / 128, 256, 0, stream>>>(x, wnode, node_b, hA);

  // --- GIN layers ---
  bf16* hcur = hA;
  bf16* hnext = hB;
  for (int l = 0; l < N_LAYERS; ++l) {
    fill_f32_kernel<<<((int)NHP + 255) / 256, 256, 0, stream>>>(agg, 0.0f, (int)NHP);
    agg_kernel<<<(N_EDGES * 32 + 255) / 256, 256, 0, stream>>>(
        eidx, eidx + N_EDGES, hcur, agg);
    gin_layer_kernel<<<M_PAD / BM, 256, 0, stream>>>(
        hcur, agg, eps, l,
        w1b + (size_t)l * HIDDEN * HIDDEN, gin_b1 + (size_t)l * HIDDEN,
        w2b + (size_t)l * HIDDEN * HIDDEN, gin_b2 + (size_t)l * HIDDEN,
        hnext);
    bf16* t = hcur; hcur = hnext; hnext = t;
  }

  // --- mean pool ---
  fill_f32_kernel<<<(N_GRAPHS * HIDDEN + 255) / 256, 256, 0, stream>>>(
      pooled, 0.0f, N_GRAPHS * HIDDEN);
  fill_f32_kernel<<<1, 256, 0, stream>>>(counts, 0.0f, N_GRAPHS);
  count_kernel<<<(N_NODES + 255) / 256, 256, 0, stream>>>(batch, counts);
  pool_sum_kernel<<<(N_NODES * 32 + 255) / 256, 256, 0, stream>>>(hcur, batch, pooled);
  mean_kernel<<<(N_GRAPHS * HIDDEN + 255) / 256, 256, 0, stream>>>(
      pooled, counts, pooled_bf);

  // --- head ---
  head_kernel<<<1, 256, 0, stream>>>(pooled_bf, ow1b, out_b1, ow2b, out_b2, out);
}
